// MoEMLPBase_42348377538842
// MI455X (gfx1250) — matrix operand order repaired and checked by reading, not compiled, not run on hardware
//
#include <hip/hip_runtime.h>
#include <math.h>

// Problem dims (fixed by reference)
#define BD 4
#define SD 1024
#define DD 1024
#define HD 4096
#define ED 8
#define KD 2
#define NTOK (BD*SD)      // 4096 tokens
#define NSLOT (NTOK*KD)   // 8192 (token, k) slots

typedef __attribute__((ext_vector_type(16))) __bf16 v16bf;
typedef __attribute__((ext_vector_type(8)))  __bf16 v8bf;
typedef __attribute__((ext_vector_type(8)))  float  v8f;

union Frag { v16bf v; v8bf h[2]; };

__device__ __forceinline__ int imin(int a, int b) { return a < b ? a : b; }

// ---------------------------------------------------------------------------
// CDNA5 async global->LDS copy (ASYNCcnt-tracked), 16B per lane.
// the LDS address is taken from the VDST VGPR; generic shared pointers carry the LDS
// offset in their low 32 bits, so a truncating cast gives the VDST value.
// ---------------------------------------------------------------------------
__device__ __forceinline__ void async_copy16(unsigned lds_off, const void* g) {
    asm volatile("global_load_async_to_lds_b128 %0, %1, off"
                 :: "v"(lds_off), "v"((unsigned long long)(uintptr_t)g)
                 : "memory");
}
__device__ __forceinline__ void wait_async_le4() {
    asm volatile("s_wait_asynccnt 0x4" ::: "memory");
}
__device__ __forceinline__ void wait_async_0() {
    asm volatile("s_wait_asynccnt 0x0" ::: "memory");
}

// ---------------------------------------------------------------------------
// K0: zero per-expert counters
// ---------------------------------------------------------------------------
__global__ void moe_init(int* cnt, float* probsum) {
    int t = threadIdx.x;
    if (t < ED) { cnt[t] = 0; probsum[t] = 0.f; }
}

// ---------------------------------------------------------------------------
// K1: router — one wave (32 lanes) per token, fp32.
// ---------------------------------------------------------------------------
__global__ __launch_bounds__(256)
void moe_router(const float* __restrict__ x, const float* __restrict__ rw,
                int* __restrict__ cnt, float* __restrict__ probsum,
                int* __restrict__ eid, int* __restrict__ pos, float* __restrict__ wgt) {
    __shared__ float s_ps[ED];
    int tid = threadIdx.x;
    if (tid < ED) s_ps[tid] = 0.f;
    __syncthreads();

    int wave = tid >> 5, lane = tid & 31;
    int t = blockIdx.x * 8 + wave;

    float acc[ED];
#pragma unroll
    for (int e = 0; e < ED; ++e) acc[e] = 0.f;

    const float* xr = x + (size_t)t * DD;
    for (int d = lane; d < DD; d += 32) {
        float xv = xr[d];
        const float* w = rw + (size_t)d * ED;
#pragma unroll
        for (int e = 0; e < ED; ++e) acc[e] += xv * w[e];
    }
#pragma unroll
    for (int off = 16; off > 0; off >>= 1) {
#pragma unroll
        for (int e = 0; e < ED; ++e) acc[e] += __shfl_down(acc[e], off, 32);
    }

    if (lane == 0) {
        float mx = acc[0];
#pragma unroll
        for (int e = 1; e < ED; ++e) mx = fmaxf(mx, acc[e]);
        float p[ED]; float s = 0.f;
#pragma unroll
        for (int e = 0; e < ED; ++e) { p[e] = __expf(acc[e] - mx); s += p[e]; }
        float inv = 1.f / s;
#pragma unroll
        for (int e = 0; e < ED; ++e) p[e] *= inv;

        int i0 = 0;
#pragma unroll
        for (int e = 1; e < ED; ++e) if (p[e] > p[i0]) i0 = e;
        int i1 = (i0 == 0) ? 1 : 0;
#pragma unroll
        for (int e = 0; e < ED; ++e) if (e != i0 && p[e] > p[i1]) i1 = e;

        float denom = p[i0] + p[i1] + 1e-8f;
        float w0 = p[i0] / denom, w1 = p[i1] / denom;

        int s0 = atomicAdd(&cnt[i0], 1);
        int s1 = atomicAdd(&cnt[i1], 1);
        int b = t * KD;
        eid[b] = i0; pos[b] = s0; wgt[b] = w0;
        eid[b + 1] = i1; pos[b + 1] = s1; wgt[b + 1] = w1;

#pragma unroll
        for (int e = 0; e < ED; ++e) atomicAdd(&s_ps[e], p[e]);
    }
    __syncthreads();
    if (tid < ED) atomicAdd(&probsum[tid], s_ps[tid]);
}

// ---------------------------------------------------------------------------
// K2: exclusive prefix sum over expert counts + aux loss
// ---------------------------------------------------------------------------
__global__ void moe_scan_aux(const int* __restrict__ cnt, const float* __restrict__ probsum,
                             int* __restrict__ off, float* __restrict__ aux_out) {
    if (threadIdx.x == 0) {
        int acc = 0; float aux = 0.f;
        const float invN = 1.f / (float)NTOK;
        const float invNK = 1.f / (float)(NTOK * KD);
#pragma unroll
        for (int e = 0; e < ED; ++e) {
            off[e] = acc; acc += cnt[e];
            aux += (probsum[e] * invN) * ((float)cnt[e] * invNK);
        }
        aux_out[0] = (float)ED * aux;
    }
}

// ---------------------------------------------------------------------------
// K3: gather token rows into expert-grouped bf16 activation matrix
// ---------------------------------------------------------------------------
__global__ __launch_bounds__(128)
void moe_gather(const float* __restrict__ x, const int* __restrict__ eid,
                const int* __restrict__ pos, const int* __restrict__ off,
                __bf16* __restrict__ xg, int* __restrict__ srow) {
    int s = blockIdx.x;
    int t = s >> 1;
    int e = eid[s];
    int row = off[e] + pos[s];
    if (threadIdx.x == 0) srow[s] = row;
    const float* src = x + (size_t)t * DD;
    __bf16* dst = xg + (size_t)row * DD;
#pragma unroll
    for (int j = 0; j < DD / 128; ++j) {
        int d = threadIdx.x + j * 128;
        dst[d] = (__bf16)src[d];
    }
}

// ---------------------------------------------------------------------------
// K3b: weight convert + transpose: W [Kd][Nd] fp32 -> WT [Nd][Kd] bf16
// (64x64 tile through LDS, coalesced both directions; run once per call)
// ---------------------------------------------------------------------------
__global__ __launch_bounds__(256)
void moe_wconv(const float* __restrict__ W, __bf16* __restrict__ WT, int Kd, int Nd) {
    const float* We = W + (size_t)blockIdx.z * Kd * Nd;
    __bf16* WTe = WT + (size_t)blockIdx.z * Kd * Nd;
    int nb = blockIdx.x * 64, kb = blockIdx.y * 64;
    __shared__ float t[64][65];
    int tid = threadIdx.x;
#pragma unroll
    for (int i = 0; i < 16; ++i) {
        int idx = i * 256 + tid;
        int r = idx >> 6, c = idx & 63;                 // r: k-local, c: n-local
        t[r][c] = We[(size_t)(kb + r) * Nd + nb + c];
    }
    __syncthreads();
#pragma unroll
    for (int i = 0; i < 16; ++i) {
        int idx = i * 256 + tid;
        int rn = idx >> 6, ck = idx & 63;               // rn: n-local, ck: k-local
        WTe[(size_t)(nb + rn) * Kd + kb + ck] = (__bf16)t[ck][rn];
    }
}

// ---------------------------------------------------------------------------
// WMMA GEMM: block = 256 threads = 8 waves (4 M x 2 N), tile M=N=128, K=32.
// A (activations) and B (transposed weights) are both bf16 [row][K] in
// global: staged into LDS with double-buffered global_load_async_to_lds_b128
// (ASYNCcnt), then consumed as ds_load_b128 fragments per ISA 7.12.2 layout.
// ---------------------------------------------------------------------------
#define TK 32
#define PAD 8
#define LPITCH (TK + PAD)   // 40 bf16 per LDS row

// K4: h = gelu(xg_e @ W_in[e] + b_in[e])  (bf16 out)
__global__ __launch_bounds__(256)
void moe_gemm1(const __bf16* __restrict__ xg, const __bf16* __restrict__ WT,
               const float* __restrict__ b_in, __bf16* __restrict__ hbuf,
               const int* __restrict__ cnt, const int* __restrict__ off) {
    int e = blockIdx.z;
    int count = cnt[e];
    int m0 = blockIdx.y * 128;
    if (m0 >= count) return;
    int n0 = blockIdx.x * 128;
    int base = off[e];
    const __bf16* Bt = WT + (size_t)e * DD * HD;   // [HD rows][DD k]

    __shared__ __bf16 sa[2][128 * LPITCH];
    __shared__ __bf16 sb[2][128 * LPITCH];

    int tid = threadIdx.x, lane = tid & 31, wv = tid >> 5;
    int wm = (wv & 3) * 32, wn = (wv >> 2) * 64;
    int lr = lane & 15, lh = lane >> 4;

    // per-thread copy slots: rows r and r+64, 16B chunk kc
    int r = tid >> 2, kc = (tid & 3) * 8;
    const __bf16* ag0 = xg + (size_t)(base + imin(m0 + r, count - 1)) * DD + kc;
    const __bf16* ag1 = xg + (size_t)(base + imin(m0 + r + 64, count - 1)) * DD + kc;
    const __bf16* bg0 = Bt + (size_t)(n0 + r) * DD + kc;
    const __bf16* bg1 = Bt + (size_t)(n0 + r + 64) * DD + kc;
    unsigned la0[2], la1[2], lb0[2], lb1[2];
#pragma unroll
    for (int b = 0; b < 2; ++b) {
        la0[b] = (unsigned)(uintptr_t)&sa[b][r * LPITCH + kc];
        la1[b] = (unsigned)(uintptr_t)&sa[b][(r + 64) * LPITCH + kc];
        lb0[b] = (unsigned)(uintptr_t)&sb[b][r * LPITCH + kc];
        lb1[b] = (unsigned)(uintptr_t)&sb[b][(r + 64) * LPITCH + kc];
    }

    v8f acc[2][4] = {};
    const int nk = DD / TK;

    // prologue: stage tile 0
    async_copy16(la0[0], ag0); async_copy16(la1[0], ag1);
    async_copy16(lb0[0], bg0); async_copy16(lb1[0], bg1);

    for (int j = 0; j < nk; ++j) {
        if (j + 1 < nk) {
            int b = (j + 1) & 1;
            size_t ko = (size_t)(j + 1) * TK;
            async_copy16(la0[b], ag0 + ko); async_copy16(la1[b], ag1 + ko);
            async_copy16(lb0[b], bg0 + ko); async_copy16(lb1[b], bg1 + ko);
            wait_async_le4();     // tile j's 4 copies (issued earlier) done
        } else {
            wait_async_0();
        }
        __syncthreads();

        const __bf16* pa = sa[j & 1];
        const __bf16* pb = sb[j & 1];
        Frag af[2], bf[4];
#pragma unroll
        for (int tm = 0; tm < 2; ++tm) {
            const __bf16* ap = pa + (wm + tm * 16 + lr) * LPITCH + lh * 8;
            af[tm].h[0] = *(const v8bf*)ap;         // K = lh*8 .. +7      -> VGPR0-3
            af[tm].h[1] = *(const v8bf*)(ap + 16);  // K = 16+lh*8 .. +7   -> VGPR4-7
        }
#pragma unroll
        for (int tn = 0; tn < 4; ++tn) {
            const __bf16* bp = pb + (wn + tn * 16 + lr) * LPITCH + lh * 8;
            bf[tn].h[0] = *(const v8bf*)bp;         // K = lh*8 .. +7
            bf[tn].h[1] = *(const v8bf*)(bp + 16);  // K = 16+lh*8 .. +7
        }
#pragma unroll
        for (int tm = 0; tm < 2; ++tm)
#pragma unroll
            for (int tn = 0; tn < 4; ++tn)
                acc[tm][tn] = __builtin_amdgcn_wmma_f32_16x16x32_bf16(
                    false, af[tm].v, false, bf[tn].v, (short)0, acc[tm][tn], false, false);
        __syncthreads();          // all reads done before buffer is re-filled
    }

    // epilogue: bias + exact GELU, store bf16
#pragma unroll
    for (int tm = 0; tm < 2; ++tm) {
#pragma unroll
        for (int tn = 0; tn < 4; ++tn) {
            int n = n0 + wn + tn * 16 + lr;
            float bias = b_in[e * HD + n];
#pragma unroll
            for (int rr = 0; rr < 8; ++rr) {
                int ml = wm + tm * 16 + lh * 8 + rr;
                if (m0 + ml < count) {
                    float v = acc[tm][tn][rr] + bias;
                    v = 0.5f * v * (1.f + erff(v * 0.70710678118f));
                    hbuf[(size_t)(base + m0 + ml) * HD + n] = (__bf16)v;
                }
            }
        }
    }
}

// K5: y = h_e @ W_out[e] + b_out[e]  (fp32 out)
__global__ __launch_bounds__(256)
void moe_gemm2(const __bf16* __restrict__ hbuf, const __bf16* __restrict__ WT,
               const float* __restrict__ b_out, float* __restrict__ ybuf,
               const int* __restrict__ cnt, const int* __restrict__ off) {
    int e = blockIdx.z;
    int count = cnt[e];
    int m0 = blockIdx.y * 128;
    if (m0 >= count) return;
    int n0 = blockIdx.x * 128;
    int base = off[e];
    const __bf16* Bt = WT + (size_t)e * HD * DD;   // [DD rows][HD k]

    __shared__ __bf16 sa[2][128 * LPITCH];
    __shared__ __bf16 sb[2][128 * LPITCH];

    int tid = threadIdx.x, lane = tid & 31, wv = tid >> 5;
    int wm = (wv & 3) * 32, wn = (wv >> 2) * 64;
    int lr = lane & 15, lh = lane >> 4;

    int r = tid >> 2, kc = (tid & 3) * 8;
    const __bf16* ag0 = hbuf + (size_t)(base + imin(m0 + r, count - 1)) * HD + kc;
    const __bf16* ag1 = hbuf + (size_t)(base + imin(m0 + r + 64, count - 1)) * HD + kc;
    const __bf16* bg0 = Bt + (size_t)(n0 + r) * HD + kc;
    const __bf16* bg1 = Bt + (size_t)(n0 + r + 64) * HD + kc;
    unsigned la0[2], la1[2], lb0[2], lb1[2];
#pragma unroll
    for (int b = 0; b < 2; ++b) {
        la0[b] = (unsigned)(uintptr_t)&sa[b][r * LPITCH + kc];
        la1[b] = (unsigned)(uintptr_t)&sa[b][(r + 64) * LPITCH + kc];
        lb0[b] = (unsigned)(uintptr_t)&sb[b][r * LPITCH + kc];
        lb1[b] = (unsigned)(uintptr_t)&sb[b][(r + 64) * LPITCH + kc];
    }

    v8f acc[2][4] = {};
    const int nk = HD / TK;

    async_copy16(la0[0], ag0); async_copy16(la1[0], ag1);
    async_copy16(lb0[0], bg0); async_copy16(lb1[0], bg1);

    for (int j = 0; j < nk; ++j) {
        if (j + 1 < nk) {
            int b = (j + 1) & 1;
            size_t ko = (size_t)(j + 1) * TK;
            async_copy16(la0[b], ag0 + ko); async_copy16(la1[b], ag1 + ko);
            async_copy16(lb0[b], bg0 + ko); async_copy16(lb1[b], bg1 + ko);
            wait_async_le4();
        } else {
            wait_async_0();
        }
        __syncthreads();

        const __bf16* pa = sa[j & 1];
        const __bf16* pb = sb[j & 1];
        Frag af[2], bf[4];
#pragma unroll
        for (int tm = 0; tm < 2; ++tm) {
            const __bf16* ap = pa + (wm + tm * 16 + lr) * LPITCH + lh * 8;
            af[tm].h[0] = *(const v8bf*)ap;
            af[tm].h[1] = *(const v8bf*)(ap + 16);
        }
#pragma unroll
        for (int tn = 0; tn < 4; ++tn) {
            const __bf16* bp = pb + (wn + tn * 16 + lr) * LPITCH + lh * 8;
            bf[tn].h[0] = *(const v8bf*)bp;
            bf[tn].h[1] = *(const v8bf*)(bp + 16);
        }
#pragma unroll
        for (int tm = 0; tm < 2; ++tm)
#pragma unroll
            for (int tn = 0; tn < 4; ++tn)
                acc[tm][tn] = __builtin_amdgcn_wmma_f32_16x16x32_bf16(
                    false, af[tm].v, false, bf[tn].v, (short)0, acc[tm][tn], false, false);
        __syncthreads();
    }

#pragma unroll
    for (int tm = 0; tm < 2; ++tm) {
#pragma unroll
        for (int tn = 0; tn < 4; ++tn) {
            int n = n0 + wn + tn * 16 + lr;
            float bias = b_out[e * DD + n];
#pragma unroll
            for (int rr = 0; rr < 8; ++rr) {
                int ml = wm + tm * 16 + lh * 8 + rr;
                if (m0 + ml < count)
                    ybuf[(size_t)(base + m0 + ml) * DD + n] = acc[tm][tn][rr] + bias;
            }
        }
    }
}

// ---------------------------------------------------------------------------
// K6: out[t,:] = w0*y[slot0,:] + w1*y[slot1,:]   (deterministic combine)
// ---------------------------------------------------------------------------
__global__ __launch_bounds__(256)
void moe_combine(const float* __restrict__ ybuf, const int* __restrict__ srow,
                 const float* __restrict__ wgt, float* __restrict__ out) {
    int t = blockIdx.x;
    int r0 = srow[t * KD], r1 = srow[t * KD + 1];
    float w0 = wgt[t * KD], w1 = wgt[t * KD + 1];
    const float* y0 = ybuf + (size_t)r0 * DD;
    const float* y1 = ybuf + (size_t)r1 * DD;
    float* o = out + (size_t)t * DD;
#pragma unroll
    for (int j = 0; j < DD / 256; ++j) {
        int d = threadIdx.x + j * 256;
        o[d] = w0 * y0[d] + w1 * y1[d];
    }
}

// ---------------------------------------------------------------------------
extern "C" void kernel_launch(void* const* d_in, const int* in_sizes, int n_in,
                              void* d_out, int out_size, void* d_ws, size_t ws_size,
                              hipStream_t stream) {
    const float* x     = (const float*)d_in[0];
    const float* rw    = (const float*)d_in[1];
    const float* W_in  = (const float*)d_in[2];
    const float* b_in  = (const float*)d_in[3];
    const float* W_out = (const float*)d_in[4];
    const float* b_out = (const float*)d_in[5];
    float* out = (float*)d_out;

    char* ws = (char*)d_ws;
    int*    cnt     = (int*)(ws + 0);
    float*  probsum = (float*)(ws + 64);
    int*    off     = (int*)(ws + 128);
    int*    eid     = (int*)(ws + 256);
    int*    pos     = (int*)(ws + 256 + 4 * (size_t)NSLOT);
    float*  wgt     = (float*)(ws + 256 + 8 * (size_t)NSLOT);
    int*    srow    = (int*)(ws + 256 + 12 * (size_t)NSLOT);
    size_t  big     = 256 + 16 * (size_t)NSLOT;                       // 131328, 16B aligned
    __bf16* xg      = (__bf16*)(ws + big);                            // 16.78 MB
    __bf16* hbuf    = (__bf16*)(ws + big + (size_t)NSLOT * DD * 2);   // 67.1 MB
    float*  ybuf    = (float*)(ws + big + (size_t)NSLOT * DD * 2
                                        + (size_t)NSLOT * HD * 2);    // 33.55 MB
    __bf16* wT      = (__bf16*)(ws + big + (size_t)NSLOT * DD * 2
                                         + (size_t)NSLOT * HD * 2
                                         + (size_t)NSLOT * DD * 4);   // 67.1 MB, reused

    moe_init<<<1, 32, 0, stream>>>(cnt, probsum);
    moe_router<<<NTOK / 8, 256, 0, stream>>>(x, rw, cnt, probsum, eid, pos, wgt);
    moe_scan_aux<<<1, 32, 0, stream>>>(cnt, probsum, off, out + (size_t)NTOK * DD);
    moe_gather<<<NSLOT, 128, 0, stream>>>(x, eid, pos, off, xg, srow);

    // W_in [E][D][H] fp32 -> wT [E][H][D] bf16, then GEMM1
    moe_wconv<<<dim3(HD / 64, DD / 64, ED), 256, 0, stream>>>(W_in, wT, DD, HD);
    moe_gemm1<<<dim3(HD / 128, NTOK / 128, ED), 256, 0, stream>>>(xg, wT, b_in, hbuf, cnt, off);

    // W_out [E][H][D] fp32 -> wT [E][D][H] bf16 (buffer reused), then GEMM2
    moe_wconv<<<dim3(DD / 64, HD / 64, ED), 256, 0, stream>>>(W_out, wT, HD, DD);
    moe_gemm2<<<dim3(DD / 128, NTOK / 128, ED), 256, 0, stream>>>(hbuf, wT, b_out, ybuf, cnt, off);

    moe_combine<<<NTOK, 256, 0, stream>>>(ybuf, srow, wgt, out);
}
